// MultiHeadAttention_16716012716386
// MI455X (gfx1250) — matrix operand order repaired and checked by reading, not compiled, not run on hardware
//
#include <hip/hip_runtime.h>

// ---------------------------------------------------------------------------
// MHA forward for gfx1250 (MI455X, wave32, WMMA 16x16x32 f16, f32 accum)
// TDM (tensor_load_to_lds) double-buffered staging in the attention core.
// GEMM K-loops: 2-stage ping/pong register pipeline (no rotation copies).
// ---------------------------------------------------------------------------

typedef __attribute__((ext_vector_type(16))) _Float16 v16h;
typedef __attribute__((ext_vector_type(8)))  _Float16 v8h;
typedef __attribute__((ext_vector_type(4)))  _Float16 v4h;
typedef __attribute__((ext_vector_type(8)))  float    v8f;
typedef __attribute__((ext_vector_type(4)))  float    v4f;
typedef __attribute__((ext_vector_type(4)))  unsigned int u32x4;
typedef __attribute__((ext_vector_type(8)))  int      i32x8;
typedef __attribute__((ext_vector_type(4)))  int      i32x4;

#define D_MODEL 1024
#define SEQ     2048
#define NHEAD   16
#define DK      64
#define BATCH   4
#define NTOK    (BATCH*SEQ)   // 8192

// D = A(16x32) * B(32x16) + C, f16 in, f32 accum
static __device__ __forceinline__ v8f wmma16(v16h a, v16h b, v8f c) {
  return __builtin_amdgcn_wmma_f32_16x16x32_f16(false, a, false, b, (short)0, c,
                                                false, false);
}

// A-fragment loader. Row-major source, `row` points at row M=lane%16.
// Lane-half g=lane/16: halves 0..15 = K=k0+16g+0..15.
static __device__ __forceinline__ v16h load_afrag(const _Float16* row, int k0, int g) {
  v8h lo = *(const v8h*)(row + k0 + 16 * g);
  v8h hi = *(const v8h*)(row + k0 + 16 * g + 8);
  v16h a;
#pragma unroll
  for (int i = 0; i < 8; ++i) { a[i] = lo[i]; a[8 + i] = hi[i]; }
  return a;
}

// ---------------------------------------------------------------------------
// Tensor Data Mover: 2D tile (f16 elements) global -> LDS.
// D# per CDNA5 ISA 8.3/8.4. tensor dims == tile dims (tiles fully in bounds),
// data_size=2B, no padding/iteration/multicast, groups 2/3 unused (2D).
// Wave-uniform; call from one wave only (EXEC ignored by TDM).
// ---------------------------------------------------------------------------
static __device__ __forceinline__ void tdm_load_2d_f16(
    unsigned lds_addr, const void* gptr,
    unsigned dim0, unsigned dim1, unsigned stride0_elems) {
  const unsigned long long ga = (unsigned long long)gptr;
  u32x4 g0;
  g0[0] = 1u;                                     // count=1 (valid user D#)
  g0[1] = lds_addr;                               // lds_addr [63:32]
  g0[2] = (unsigned)ga;                           // global_addr lo
  g0[3] = (unsigned)(ga >> 32) | (2u << 30);      // global_addr hi | type=2
  i32x8 g1;
  g1[0] = (int)(1u << 16);                        // data_size=1 (2 bytes)
  g1[1] = (int)((dim0 & 0xFFFFu) << 16);          // tensor_dim0[15:0]
  g1[2] = (int)(((dim0 >> 16) & 0xFFFFu) |        // tensor_dim0[31:16]
                ((dim1 & 0xFFFFu) << 16));        // tensor_dim1[15:0]
  g1[3] = (int)(((dim1 >> 16) & 0xFFFFu) |        // tensor_dim1[31:16]
                ((dim0 & 0xFFFFu) << 16));        // tile_dim0
  g1[4] = (int)(dim1 & 0xFFFFu);                  // tile_dim1 (tile_dim2=0)
  g1[5] = (int)stride0_elems;                     // tensor_dim0_stride[31:0]
  g1[6] = 0;                                      // stride0 hi | stride1 lo
  g1[7] = 0;                                      // stride1 hi
  const i32x4 z4 = {0, 0, 0, 0};
#if defined(__clang_major__) && (__clang_major__ >= 23)
  const i32x8 z8 = {};
  __builtin_amdgcn_tensor_load_to_lds(g0, g1, z4, z4, z8, 0);
#else
  __builtin_amdgcn_tensor_load_to_lds(g0, g1, z4, z4, 0);
#endif
}

// ---------------------------------------------------------------------------
// f32 -> f16 conversion (x and the four weight matrices)
// ---------------------------------------------------------------------------
__global__ __launch_bounds__(256)
void mha_cvt_f16(const float* __restrict__ s, _Float16* __restrict__ d, int n) {
  int i = (blockIdx.x * 256 + threadIdx.x) * 4;
  if (i + 3 < n) {
    v4f f = *(const v4f*)(s + i);
    v4h o;
#pragma unroll
    for (int j = 0; j < 4; ++j) o[j] = (_Float16)f[j];
    *(v4h*)(d + i) = o;
  } else {
    for (; i < n; ++i) d[i] = (_Float16)s[i];
  }
}

// ---------------------------------------------------------------------------
// Fused QKV projection:  y[m,e] = sum_d X[m,d]*W[e,d] + b[e]
// grid (NTOK/64, D_MODEL/64, 3), block 128 (4 waves, each a 16Mx64N strip).
// 2-stage ping/pong pipeline: stage-A fragments compute while stage-B
// fragments load, and vice versa — no register rotation movs.
// z=0 -> Q [b,h,s,dk], z=1 -> K [b,h,s,dk], z=2 -> V transposed [b,h,dk,s].
// ---------------------------------------------------------------------------
__global__ __launch_bounds__(128)
void mha_qkv_proj(const _Float16* __restrict__ X,
                  const _Float16* __restrict__ Wq, const _Float16* __restrict__ Wk,
                  const _Float16* __restrict__ Wv,
                  const float* __restrict__ bq, const float* __restrict__ bk,
                  const float* __restrict__ bv,
                  _Float16* __restrict__ Q, _Float16* __restrict__ Ko,
                  _Float16* __restrict__ Vt) {
  const int z = blockIdx.z;
  const _Float16* W = (z == 0) ? Wq : ((z == 1) ? Wk : Wv);
  const float* bias = (z == 0) ? bq : ((z == 1) ? bk : bv);

  const int lane = threadIdx.x & 31, wv = threadIdx.x >> 5;
  const int g = lane >> 4, ln = lane & 15;
  const int mm = blockIdx.x * 64 + wv * 16;   // M tile base (token rows)
  const int n0 = blockIdx.y * 64;             // N base (4 tiles of 16)

  const _Float16* arow = X + (mm + ln) * D_MODEL;
  const _Float16* brow[4];
#pragma unroll
  for (int t = 0; t < 4; ++t) brow[t] = W + (n0 + t * 16 + ln) * D_MODEL + 16 * g;

  // prologue: stage-A fragments for k = 0
  v16h a0 = load_afrag(arow, 0, g);
  v16h bf0[4], bf1[4];
#pragma unroll
  for (int t = 0; t < 4; ++t) bf0[t] = *(const v16h*)(brow[t]);

  v8f acc[4] = {};
  for (int k0 = 0; k0 < D_MODEL; k0 += 64) {
    // stage B loads (k0+32) while stage A computes
    v16h a1 = load_afrag(arow, k0 + 32, g);
#pragma unroll
    for (int t = 0; t < 4; ++t) bf1[t] = *(const v16h*)(brow[t] + k0 + 32);
    __builtin_prefetch((const void*)(arow + k0 + 192), 0, 0);
#pragma unroll
    for (int t = 0; t < 4; ++t) acc[t] = wmma16(a0, bf0[t], acc[t]);

    // stage A loads (k0+64, wraps on last iter) while stage B computes
    const int k2 = (k0 + 64) & (D_MODEL - 1);
    a0 = load_afrag(arow, k2, g);
#pragma unroll
    for (int t = 0; t < 4; ++t) bf0[t] = *(const v16h*)(brow[t] + k2);
#pragma unroll
    for (int t = 0; t < 4; ++t) acc[t] = wmma16(a1, bf1[t], acc[t]);
  }

  const int b = mm >> 11, s0 = mm & (SEQ - 1);
#pragma unroll
  for (int t = 0; t < 4; ++t) {
    const int e = n0 + t * 16 + ln;           // output feature (per lane)
    const float bb = bias[e];
    const int h = e >> 6, d = e & 63;
    if (z < 2) {
      _Float16* O = (z == 0) ? Q : Ko;        // [b,h,s,dk]
#pragma unroll
      for (int r = 0; r < 8; ++r) {
        const int s = s0 + r + 8 * g;         // C layout: M = r + 8g
        O[(((b * NHEAD + h) * SEQ + s) << 6) + d] = (_Float16)(acc[t][r] + bb);
      }
    } else {                                  // V transposed: [b,h,dk,s]
      v8h pk;
#pragma unroll
      for (int r = 0; r < 8; ++r) pk[r] = (_Float16)(acc[t][r] + bb);
      *(v8h*)(Vt + (((b * NHEAD + h) * DK + d) << 11) + s0 + 8 * g) = pk;
    }
  }
}

// ---------------------------------------------------------------------------
// Flash attention: grid (SEQ/64, B*H), block 128. Wave w owns 16 queries.
// K/V^T 32-key tiles staged into ping-pong LDS buffers by the Tensor Data
// Mover (wave 0 issues, all waves s_wait_tensorcnt + barrier). Next block's
// DMA overlaps the current block's 8 WMMAs + online softmax.
// ---------------------------------------------------------------------------
__global__ __launch_bounds__(128)
void mha_flash_attn(const _Float16* __restrict__ Q, const _Float16* __restrict__ K,
                    const _Float16* __restrict__ Vt, _Float16* __restrict__ U) {
  __shared__ __attribute__((aligned(64))) _Float16 kbuf[2][32 * 64];    // [k][d]
  __shared__ __attribute__((aligned(64))) _Float16 vbuf[2][64 * 32];    // [d][k]
  __shared__ __attribute__((aligned(32))) _Float16 pbuf[4][16 * 32];    // per-wave [q][k]
  __shared__ __attribute__((aligned(32))) _Float16 obuf[4][16 * 64];    // per-wave [q][d]

  const int tid = threadIdx.x;
  const int lane = tid & 31, wv = tid >> 5;
  const int g = lane >> 4, ln = lane & 15;
  const int bh = blockIdx.y;
  const int q0 = blockIdx.x * 64 + wv * 16;

  const _Float16* Kb = K + bh * SEQ * DK;
  const _Float16* Vb = Vt + bh * DK * SEQ;

  // Kick off DMA of the first key block while we load Q fragments.
  if (wv == 0) {
    tdm_load_2d_f16((unsigned)(uintptr_t)&kbuf[0][0], Kb, DK, 32, DK);
    tdm_load_2d_f16((unsigned)(uintptr_t)&vbuf[0][0], Vb, 32, DK, SEQ);
  }

  // Q^T B-fragments, kept in registers for the whole kernel (2 x 8 VGPRs)
  const _Float16* qrow = Q + (bh * SEQ + q0 + ln) * DK;
  const v16h qb0 = *(const v16h*)(qrow + 0 + 16 * g);
  const v16h qb1 = *(const v16h*)(qrow + 32 + 16 * g);

  v8f acc[4] = {};                 // out^T tiles, d in [16t,16t+16), q = ln
  float m = -1e30f, l = 0.f;
  const float C1 = 0.125f * 1.44269504f;   // 1/sqrt(dk) * log2(e)

  __builtin_amdgcn_s_wait_tensorcnt(0);
  __syncthreads();

  for (int kb = 0; kb < SEQ; kb += 32) {
    const int cur = (kb >> 5) & 1;
    // DMA the next block into the other buffer; overlaps this block's math.
    if (wv == 0 && kb + 32 < SEQ) {
      tdm_load_2d_f16((unsigned)(uintptr_t)&kbuf[cur ^ 1][0],
                      Kb + (kb + 32) * DK, DK, 32, DK);
      tdm_load_2d_f16((unsigned)(uintptr_t)&vbuf[cur ^ 1][0],
                      Vb + (kb + 32), 32, DK, SEQ);
    }

    // S^T tiles: (M = key local, N = query)
    v8f st[2];
#pragma unroll
    for (int t2 = 0; t2 < 2; ++t2) {
      const _Float16* krow = &kbuf[cur][(t2 * 16 + ln) * DK];
      v8f s_ = {};
      s_ = wmma16(load_afrag(krow, 0, g), qb0, s_);
      s_ = wmma16(load_afrag(krow, 32, g), qb1, s_);
      st[t2] = s_;
    }

    // online softmax (per query = per lane column): 8 in-lane + xor-16
    float tmax = st[0][0];
#pragma unroll
    for (int r = 0; r < 8; ++r) {
      tmax = fmaxf(tmax, st[0][r]);
      tmax = fmaxf(tmax, st[1][r]);
    }
    tmax = fmaxf(tmax, __shfl_xor(tmax, 16, 32));
    const float mn = fmaxf(m, tmax);
    const float alpha = __builtin_amdgcn_exp2f((m - mn) * C1);
    const float mc = mn * C1;
    float rsum = 0.f;
    v8h pk0, pk1;
#pragma unroll
    for (int r = 0; r < 8; ++r) {
      float p0 = __builtin_amdgcn_exp2f(st[0][r] * C1 - mc);
      float p1 = __builtin_amdgcn_exp2f(st[1][r] * C1 - mc);
      rsum += p0 + p1;
      pk0[r] = (_Float16)p0;
      pk1[r] = (_Float16)p1;
    }
    rsum += __shfl_xor(rsum, 16, 32);
    l = l * alpha + rsum;
    m = mn;

    // P^T -> per-wave LDS [q][k], reload as B-fragment. Same-wave DS ops are
    // in-order; only a compiler reordering fence is needed (no s_barrier).
    *(v8h*)&pbuf[wv][ln * 32 + 0 + 8 * g] = pk0;
    *(v8h*)&pbuf[wv][ln * 32 + 16 + 8 * g] = pk1;
    asm volatile("" ::: "memory");

#pragma unroll
    for (int t = 0; t < 4; ++t)
#pragma unroll
      for (int r = 0; r < 8; ++r) acc[t][r] *= alpha;

    const v16h pb = *(const v16h*)&pbuf[wv][ln * 32 + 16 * g];
#pragma unroll
    for (int t = 0; t < 4; ++t) {
      const _Float16* vrow = &vbuf[cur][(t * 16 + ln) * 32];
      acc[t] = wmma16(load_afrag(vrow, 0, g), pb, acc[t]);
    }

    // Next buffer must be DMA-complete (wave 0) and everyone must be done
    // reading the current buffer before it is overwritten next iteration.
    __builtin_amdgcn_s_wait_tensorcnt(0);
    __syncthreads();
  }

  // normalize, transpose via LDS, coalesced 64B stores to U [tok][1024]
  const float inv = 1.f / l;
#pragma unroll
  for (int t = 0; t < 4; ++t) {
    v8h ok;
#pragma unroll
    for (int r = 0; r < 8; ++r) ok[r] = (_Float16)(acc[t][r] * inv);
    *(v8h*)&obuf[wv][ln * 64 + t * 16 + 8 * g] = ok;
  }
  __syncthreads();
  {
    const int q = lane >> 1, half = lane & 1;
    const _Float16* src = &obuf[wv][q * 64 + half * 32];
    const int b = bh >> 4, h = bh & 15;
    _Float16* dst = U + (((long)b * SEQ + q0 + q) << 10) + h * 64 + half * 32;
    *(v16h*)dst = *(const v16h*)src;
    *(v16h*)(dst + 16) = *(const v16h*)(src + 16);
  }
}

// ---------------------------------------------------------------------------
// Output projection: out[m,e] = sum_d U[m,d]*Wo[e,d] + bo[e]   (f32 out)
// grid (NTOK/64, D_MODEL/64), block 128. 2-stage ping/pong pipeline.
// ---------------------------------------------------------------------------
__global__ __launch_bounds__(128)
void mha_o_proj(const _Float16* __restrict__ U, const _Float16* __restrict__ Wo,
                const float* __restrict__ bo, float* __restrict__ out) {
  const int lane = threadIdx.x & 31, wv = threadIdx.x >> 5;
  const int g = lane >> 4, ln = lane & 15;
  const int mm = blockIdx.x * 64 + wv * 16;
  const int n0 = blockIdx.y * 64;

  const _Float16* arow = U + (mm + ln) * D_MODEL;
  const _Float16* brow[4];
#pragma unroll
  for (int t = 0; t < 4; ++t) brow[t] = Wo + (n0 + t * 16 + ln) * D_MODEL + 16 * g;

  v16h a0 = load_afrag(arow, 0, g);
  v16h bf0[4], bf1[4];
#pragma unroll
  for (int t = 0; t < 4; ++t) bf0[t] = *(const v16h*)(brow[t]);

  v8f acc[4] = {};
  for (int k0 = 0; k0 < D_MODEL; k0 += 64) {
    v16h a1 = load_afrag(arow, k0 + 32, g);
#pragma unroll
    for (int t = 0; t < 4; ++t) bf1[t] = *(const v16h*)(brow[t] + k0 + 32);
    __builtin_prefetch((const void*)(arow + k0 + 192), 0, 0);
#pragma unroll
    for (int t = 0; t < 4; ++t) acc[t] = wmma16(a0, bf0[t], acc[t]);

    const int k2 = (k0 + 64) & (D_MODEL - 1);
    a0 = load_afrag(arow, k2, g);
#pragma unroll
    for (int t = 0; t < 4; ++t) bf0[t] = *(const v16h*)(brow[t] + k2);
#pragma unroll
    for (int t = 0; t < 4; ++t) acc[t] = wmma16(a1, bf1[t], acc[t]);
  }
#pragma unroll
  for (int t = 0; t < 4; ++t) {
    const int e = n0 + t * 16 + ln;
    const float bb = bo[e];
#pragma unroll
    for (int r = 0; r < 8; ++r)
      out[(long)(mm + r + 8 * g) * D_MODEL + e] = acc[t][r] + bb;   // 64B/row coalesced
  }
}

// ---------------------------------------------------------------------------
extern "C" void kernel_launch(void* const* d_in, const int* in_sizes, int n_in,
                              void* d_out, int out_size, void* d_ws, size_t ws_size,
                              hipStream_t stream) {
  (void)in_sizes; (void)n_in; (void)out_size; (void)ws_size;
  const float* x  = (const float*)d_in[0];
  const float* Wq = (const float*)d_in[1];
  const float* bq = (const float*)d_in[2];
  const float* Wk = (const float*)d_in[3];
  const float* bk = (const float*)d_in[4];
  const float* Wv = (const float*)d_in[5];
  const float* bv = (const float*)d_in[6];
  const float* Wo = (const float*)d_in[7];
  const float* bo = (const float*)d_in[8];

  char* ws = (char*)d_ws;
  _Float16* Xh  = (_Float16*)(ws);                          // 16 MB
  _Float16* Wqh = (_Float16*)(ws + (size_t)(16u << 20));    //  2 MB each
  _Float16* Wkh = (_Float16*)(ws + (size_t)(18u << 20));
  _Float16* Wvh = (_Float16*)(ws + (size_t)(20u << 20));
  _Float16* Woh = (_Float16*)(ws + (size_t)(22u << 20));
  _Float16* Qb  = (_Float16*)(ws + (size_t)(24u << 20));    // 16 MB
  _Float16* Kb  = (_Float16*)(ws + (size_t)(40u << 20));    // 16 MB
  _Float16* Vt  = (_Float16*)(ws + (size_t)(56u << 20));    // 16 MB
  _Float16* U   = (_Float16*)(ws + (size_t)(72u << 20));    // 16 MB (88 MB total)

  const int nX = NTOK * D_MODEL, nW = D_MODEL * D_MODEL;
  mha_cvt_f16<<<dim3((nX / 4 + 255) / 256), 256, 0, stream>>>(x,  Xh,  nX);
  mha_cvt_f16<<<dim3((nW / 4 + 255) / 256), 256, 0, stream>>>(Wq, Wqh, nW);
  mha_cvt_f16<<<dim3((nW / 4 + 255) / 256), 256, 0, stream>>>(Wk, Wkh, nW);
  mha_cvt_f16<<<dim3((nW / 4 + 255) / 256), 256, 0, stream>>>(Wv, Wvh, nW);
  mha_cvt_f16<<<dim3((nW / 4 + 255) / 256), 256, 0, stream>>>(Wo, Woh, nW);

  mha_qkv_proj<<<dim3(NTOK / 64, D_MODEL / 64, 3), 128, 0, stream>>>(
      Xh, Wqh, Wkh, Wvh, bq, bk, bv, Qb, Kb, Vt);

  mha_flash_attn<<<dim3(SEQ / 64, BATCH * NHEAD), 128, 0, stream>>>(Qb, Kb, Vt, U);

  mha_o_proj<<<dim3(NTOK / 64, D_MODEL / 64), 128, 0, stream>>>(
      U, Woh, bo, (float*)d_out);
}
